// Attention_39797166965273
// MI455X (gfx1250) — matrix operand order repaired and checked by reading, not compiled, not run on hardware
//
#include <hip/hip_runtime.h>

// MI455X / gfx1250: wave32, WMMA f32<=f16 16x16x32 path.
// Layout conventions follow CDNA5 ISA:
//  A (16x32 f16): lane m=lane&15, h=lane>>4; a[0..7]=A[m][h*8+i], a[8..15]=A[m][16+h*8+i]
//  B (32x16 f16): lane n=lane&15, kh=lane>>4; b[0..7]=B[kh*8+i][n], b[8..15]=B[16+kh*8+i][n]
//  C/D (16x16 f32): lane n=lane&15, h=lane>>4; c[i]=C[i+8h][n]

typedef __attribute__((ext_vector_type(16))) _Float16 v16h;
typedef __attribute__((ext_vector_type(8)))  _Float16 v8h;
typedef __attribute__((ext_vector_type(8)))  float    v8f;
typedef __attribute__((ext_vector_type(4)))  float    v4f;

namespace {
constexpr int BD    = 256;        // B == N == D == 256
constexpr int NROWS = BD * BD;    // 65536 flattened (b,n) rows
}

__device__ __forceinline__ v8f wmma16(v16h a, v16h b, v8f c) {
  return __builtin_amdgcn_wmma_f32_16x16x32_f16(false, a, false, b, (short)0, c,
                                                false, false);
}

__device__ __forceinline__ v8f zero8() {
  v8f z;
#pragma unroll
  for (int i = 0; i < 8; ++i) z[i] = 0.0f;
  return z;
}

// Load 8 contiguous elements (f32 or f16 in memory) into halves a[lo..lo+7].
template <typename TIN>
__device__ __forceinline__ void load8(const TIN* p, v16h& a, int lo) {
  if constexpr (sizeof(TIN) == 2) {
    v8h v = *(const v8h*)p;
#pragma unroll
    for (int i = 0; i < 8; ++i) a[lo + i] = v[i];
  } else {
    v4f v0 = *(const v4f*)p;
    v4f v1 = *(const v4f*)(p + 4);
#pragma unroll
    for (int i = 0; i < 4; ++i) {
      a[lo + i]     = (_Float16)v0[i];
      a[lo + 4 + i] = (_Float16)v1[i];
    }
  }
}

// ---------------- W fp32 -> fp16 (all four weight matrices) ----------------
__global__ void __launch_bounds__(256)
convert_w_kernel(const float* __restrict__ W1, const float* __restrict__ W2,
                 const float* __restrict__ W3, const float* __restrict__ W4,
                 _Float16* __restrict__ Wh) {
  const int idx = blockIdx.x * blockDim.x + threadIdx.x;  // 0 .. 4*65536-1
  const int m = idx >> 16;
  const int o = idx & 0xFFFF;
  const float* src = (m == 0) ? W1 : (m == 1) ? W2 : (m == 2) ? W3 : W4;
  Wh[idx] = (_Float16)src[o];
}

// ---------------- Y = X @ W^T + b  (M=65536, K=N=256) ----------------------
// Block: 128 threads = 4 waves; block tile 64(M) x 64(E); wave tile 16 x 64.
template <typename TIN, typename TOUT>
__global__ void __launch_bounds__(128)
gemm_bias_kernel(const TIN* __restrict__ X, const _Float16* __restrict__ Wh,
                 const float* __restrict__ bias, TOUT* __restrict__ Y) {
  const int lane = threadIdx.x & 31;
  const int wave = threadIdx.x >> 5;
  const int m0   = blockIdx.x * 64 + wave * 16;
  const int e0   = blockIdx.y * 64;
  const int n16  = lane & 15;
  const int h    = lane >> 4;

  // Hoist all 8 A fragments for this wave's 16 rows (K = 256 = 8 * 32).
  const TIN* xrow = X + (size_t)(m0 + n16) * BD;
  v16h afr[8];
#pragma unroll
  for (int kc = 0; kc < 8; ++kc) {
    const int k0 = kc * 32 + h * 8;
    load8<TIN>(xrow + k0, afr[kc], 0);
    load8<TIN>(xrow + k0 + 16, afr[kc], 8);
  }

  v8f acc[4];
#pragma unroll
  for (int t = 0; t < 4; ++t) acc[t] = zero8();

#pragma unroll
  for (int kc = 0; kc < 8; ++kc) {
#pragma unroll
    for (int t = 0; t < 4; ++t) {
      // B[k][j] = W[e0+t*16+j][k]  ->  b[i] = W[e0+t*16+n16][kc*32 + h*8 + i], b[8+i] = W[..][kc*32 + 16 + h*8 + i]
      const _Float16* wrow =
          Wh + (size_t)(e0 + t * 16 + n16) * BD + kc * 32 + h * 8;
      v16h bfr;
      load8<_Float16>(wrow, bfr, 0);
      load8<_Float16>(wrow + 16, bfr, 8);
      acc[t] = wmma16(afr[kc], bfr, acc[t]);
    }
  }

#pragma unroll
  for (int t = 0; t < 4; ++t) {
    const float bv = bias[e0 + t * 16 + n16];
#pragma unroll
    for (int i = 0; i < 8; ++i) {
      const int r = m0 + i + 8 * h;
      Y[(size_t)r * BD + e0 + t * 16 + n16] = (TOUT)(acc[t][i] + bv);
    }
  }
}

// -------- per-n statistics over (b, e): mean and rsqrt(var + eps) ----------
template <typename T>
__global__ void __launch_bounds__(256)
stats_kernel(const T* __restrict__ Y, float* __restrict__ mean,
             float* __restrict__ rstd) {
  __shared__ float ss[256];
  __shared__ float sq[256];
  const int n = blockIdx.x;
  const int t = threadIdx.x;
  float s = 0.0f, q = 0.0f;
  for (int b = 0; b < BD; ++b) {
    const float v = (float)Y[((size_t)b * BD + n) * BD + t];
    s += v;
    q += v * v;
  }
  ss[t] = s;
  sq[t] = q;
  __syncthreads();
  for (int o = 128; o > 0; o >>= 1) {
    if (t < o) { ss[t] += ss[t + o]; sq[t] += sq[t + o]; }
    __syncthreads();
  }
  if (t == 0) {
    const float inv = 1.0f / (float)(BD * BD);
    const float m   = ss[0] * inv;
    const float var = sq[0] * inv - m * m;
    mean[n] = m;
    rstd[n] = rsqrtf(var + 1e-5f);
  }
}

// -------- in-place normalize (per n) + affine + ReLU -----------------------
template <typename T>
__global__ void __launch_bounds__(256)
bn_relu_kernel(T* __restrict__ Y, const float* __restrict__ mean,
               const float* __restrict__ rstd, const float* __restrict__ gamma,
               const float* __restrict__ beta) {
  const int idx = blockIdx.x * blockDim.x + threadIdx.x;  // < 16777216
  const int n   = (idx >> 8) & 255;                        // idx = b*65536 + n*256 + e
  float v = (float)Y[idx];
  v = (v - mean[n]) * rstd[n] * gamma[n] + beta[n];
  Y[idx] = (T)fmaxf(v, 0.0f);
}

// -------- fused attention: O = softmax(x1 x2^T / 16) x3, per batch ---------
// Block: 64 threads = 2 waves; each wave owns 16 query rows of one batch.
__global__ void __launch_bounds__(64)
attention_kernel(const _Float16* __restrict__ x1, const _Float16* __restrict__ x2,
                 const _Float16* __restrict__ x3, _Float16* __restrict__ O) {
  __shared__ float S[2][16][264];   // per-wave 16x256 score slab (+pad)
  __shared__ float rinv[2][16];     // per-row 1/rowsum

  const int lane  = threadIdx.x & 31;
  const int wave  = threadIdx.x >> 5;
  const int batch = blockIdx.y;
  const int n0    = blockIdx.x * 32 + wave * 16;
  const int n16   = lane & 15;
  const int h     = lane >> 4;
  const size_t base = (size_t)batch * BD * BD;

  // Hoist A fragments of x1 rows n0..n0+15 for all K chunks.
  v16h afr[8];
  const _Float16* arow = x1 + base + (size_t)(n0 + n16) * BD;
#pragma unroll
  for (int kc = 0; kc < 8; ++kc) {
    const int k0 = kc * 32 + h * 8;
    load8<_Float16>(arow + k0, afr[kc], 0);
    load8<_Float16>(arow + k0 + 16, afr[kc], 8);
  }

  // Phase 1: S = (x1 @ x2^T) * (1/16)
  for (int mt = 0; mt < 16; ++mt) {
    v8f acc = zero8();
    // B[k][j] = x2[mt*16+j][k] -> two 8-half loads per lane at h*8 and 16+h*8
    const _Float16* brow = x2 + base + (size_t)(mt * 16 + n16) * BD;
#pragma unroll
    for (int kc = 0; kc < 8; ++kc) {
      v16h bfr;
      load8<_Float16>(brow + kc * 32 + h * 8, bfr, 0);
      load8<_Float16>(brow + kc * 32 + h * 8 + 16, bfr, 8);
      acc = wmma16(afr[kc], bfr, acc);
    }
#pragma unroll
    for (int i = 0; i < 8; ++i)
      S[wave][i + 8 * h][mt * 16 + n16] = acc[i] * 0.0625f;
  }
  __syncthreads();

  // Phase 2: row softmax; exp() stored back into S, rowsum inverted.
  for (int r = 0; r < 16; ++r) {
    float mx = -3.0e38f;
    for (int c = lane; c < BD; c += 32) mx = fmaxf(mx, S[wave][r][c]);
#pragma unroll
    for (int o = 16; o > 0; o >>= 1) mx = fmaxf(mx, __shfl_xor(mx, o, 32));
    float sm = 0.0f;
    for (int c = lane; c < BD; c += 32) {
      const float e = __expf(S[wave][r][c] - mx);
      S[wave][r][c] = e;
      sm += e;
    }
#pragma unroll
    for (int o = 16; o > 0; o >>= 1) sm += __shfl_xor(sm, o, 32);
    if (lane == 0) rinv[wave][r] = 1.0f / sm;
  }
  __syncthreads();

  // Phase 3: O = P @ x3 (P rebuilt from S * rinv as f16 A fragments)
  const float rs = rinv[wave][n16];  // scale for A row m = lane&15
  for (int dt = 0; dt < 16; ++dt) {
    v8f acc = zero8();
#pragma unroll
    for (int kc = 0; kc < 8; ++kc) {
      v16h pa;
      const int k0 = kc * 32 + h * 16;
#pragma unroll
      for (int i = 0; i < 8; ++i) {
        pa[i]     = (_Float16)(S[wave][n16][k0 + i] * rs);
        pa[8 + i] = (_Float16)(S[wave][n16][k0 + 8 + i] * rs);
      }
      // B[k][j] = x3[k][dt*16+j]
      const _Float16* brow =
          x3 + base + (size_t)(kc * 32 + h * 16) * BD + dt * 16 + n16;
      v16h bfr;
#pragma unroll
      for (int i = 0; i < 16; ++i) bfr[i] = brow[(size_t)i * BD];
      acc = wmma16(pa, bfr, acc);
    }
#pragma unroll
    for (int i = 0; i < 8; ++i)
      O[base + (size_t)(n0 + i + 8 * h) * BD + dt * 16 + n16] = (_Float16)acc[i];
  }
}

// ---------------------------------------------------------------------------
extern "C" void kernel_launch(void* const* d_in, const int* in_sizes, int n_in,
                              void* d_out, int out_size, void* d_ws, size_t ws_size,
                              hipStream_t stream) {
  (void)in_sizes; (void)n_in; (void)out_size; (void)ws_size;
  const float* x     = (const float*)d_in[0];
  const float* b1    = (const float*)d_in[2];
  const float* b2    = (const float*)d_in[4];
  const float* b3    = (const float*)d_in[6];
  const float* b4    = (const float*)d_in[8];
  const float* gamma = (const float*)d_in[9];
  const float* beta  = (const float*)d_in[10];

  // Workspace carve-up (~129 MB).
  char* p = (char*)d_ws;
  _Float16* Wh = (_Float16*)p; p += (size_t)4 * 65536 * sizeof(_Float16);
  _Float16* X1 = (_Float16*)p; p += (size_t)NROWS * BD * sizeof(_Float16);
  _Float16* X2 = (_Float16*)p; p += (size_t)NROWS * BD * sizeof(_Float16);
  _Float16* X3 = (_Float16*)p; p += (size_t)NROWS * BD * sizeof(_Float16);
  _Float16* Oh = (_Float16*)p; p += (size_t)NROWS * BD * sizeof(_Float16);
  float* mean  = (float*)p;    p += (size_t)4 * 256 * sizeof(float);
  float* rstd  = (float*)p;    p += (size_t)4 * 256 * sizeof(float);

  float* Z = (float*)d_out;

  // 0) weights to f16
  convert_w_kernel<<<1024, 256, 0, stream>>>(
      (const float*)d_in[1], (const float*)d_in[3],
      (const float*)d_in[5], (const float*)d_in[7], Wh);

  // 1) three projection GEMMs (unnormalized, f16)
  const dim3 gg(NROWS / 64, BD / 64);
  gemm_bias_kernel<float, _Float16><<<gg, 128, 0, stream>>>(x, Wh,           b1, X1);
  gemm_bias_kernel<float, _Float16><<<gg, 128, 0, stream>>>(x, Wh + 65536,   b2, X2);
  gemm_bias_kernel<float, _Float16><<<gg, 128, 0, stream>>>(x, Wh + 131072,  b3, X3);

  // 2) BN(+ReLU) per n, in place
  stats_kernel<_Float16><<<256, 256, 0, stream>>>(X1, mean + 0,   rstd + 0);
  stats_kernel<_Float16><<<256, 256, 0, stream>>>(X2, mean + 256, rstd + 256);
  stats_kernel<_Float16><<<256, 256, 0, stream>>>(X3, mean + 512, rstd + 512);
  bn_relu_kernel<_Float16><<<65536, 256, 0, stream>>>(X1, mean + 0,   rstd + 0,   gamma, beta);
  bn_relu_kernel<_Float16><<<65536, 256, 0, stream>>>(X2, mean + 256, rstd + 256, gamma, beta);
  bn_relu_kernel<_Float16><<<65536, 256, 0, stream>>>(X3, mean + 512, rstd + 512, gamma, beta);

  // 3) fused attention per batch
  attention_kernel<<<dim3(BD / 32, BD), 64, 0, stream>>>(X1, X2, X3, Oh);

  // 4) output projection + BN(+ReLU) in place on d_out
  gemm_bias_kernel<_Float16, float><<<gg, 128, 0, stream>>>(Oh, Wh + 196608, b4, Z);
  stats_kernel<float><<<256, 256, 0, stream>>>(Z, mean + 768, rstd + 768);
  bn_relu_kernel<float><<<65536, 256, 0, stream>>>(Z, mean + 768, rstd + 768, gamma, beta);
}
